// Causal_v_69054484185473
// MI455X (gfx1250) — matrix operand order repaired and checked by reading, not compiled, not run on hardware
//
#include <hip/hip_runtime.h>
#include <hip/hip_bf16.h>

// ---------------------------------------------------------------------------
// MI455X / gfx1250 fused dictionary-attention pipeline.
//
// Precision: bf16 operands, f32 WMMA accumulation (v_wmma_f32_16x16x32_bf16),
// f32 softmax statistics, f32 output. Compute-bound (~0.46 TFLOP vs ~10us of
// HBM traffic at 23.3 TB/s), so all GEMMs go through WMMA; the [16384 x 4096]
// attention matrix never touches HBM (32 rows x 4096 cols bf16, padded, in
// the 320KB/WGP LDS). Working set (h 32MB, dz 8MB, dicT 16MB) is L2-resident
// (192MB global L2), so fragment re-reads hit L2, not HBM.
//
// LDS layout uses a padded row stride (4096+8 bf16 = 8208B) so that
// lane-indexed-by-row accesses (stage-3 A-fragments) spread across all 64
// banks instead of aliasing to one bank group.
// ---------------------------------------------------------------------------

typedef __attribute__((ext_vector_type(16))) __bf16 v16bf;
typedef __attribute__((ext_vector_type(8)))  float  v8f;
typedef __attribute__((ext_vector_type(4)))  int    vi4;

#if __has_builtin(__builtin_amdgcn_global_load_async_to_lds_b128)
#define HAVE_ASYNC_LDS 1
#else
#define HAVE_ASYNC_LDS 0
#endif

union Frag {
    uint4 u[2];   // two 16B chunks, loaded per the ISA 16-bit fragment layout
    v16bf v;
};

static __device__ __forceinline__ unsigned short f2bf(float f) {
    unsigned u = __float_as_uint(f);
    u += 0x7FFFu + ((u >> 16) & 1u);        // round-to-nearest-even
    return (unsigned short)(u >> 16);
}
static __device__ __forceinline__ float bf2f(unsigned short h) {
    return __uint_as_float(((unsigned)h) << 16);
}
static __device__ __forceinline__ float lo_bf(unsigned u) {
    return __uint_as_float(u << 16);
}
static __device__ __forceinline__ float hi_bf(unsigned u) {
    return __uint_as_float(u & 0xFFFF0000u);
}
static __device__ __forceinline__ unsigned pack2(float a, float b) {
    return (unsigned)f2bf(a) | ((unsigned)f2bf(b) << 16);
}

static __device__ __forceinline__ v8f wmma_bf16(const Frag& a, const Frag& b, v8f c) {
    // D = A(16x32) * B(32x16) + C(16x16), f32 accumulate
    return __builtin_amdgcn_wmma_f32_16x16x32_bf16(
        /*neg_a=*/false, a.v, /*neg_b=*/false, b.v,
        /*c_mod=*/(short)0, c, /*reuse_a=*/false, /*reuse_b=*/false);
}

// A-fragment (16x32, MxK) from row-major [rows, ld] bf16, K contiguous.
// Lane l: row = base_row + (l&15); two 8-elem runs at K = k0+8h and k0+16+8h.
static __device__ __forceinline__ Frag load_afrag(const unsigned short* rowp, int k0, int hf) {
    Frag a;
    const unsigned short* p = rowp + k0 + 8 * hf;
    a.u[0] = *(const uint4*)(p);
    a.u[1] = *(const uint4*)(p + 16);
    return a;
}
// B-fragment (32x16, KxN) from row-major [N, ld] bf16 (i.e. B^T, K contiguous).
// Lane l: col = base_col + (l&15); two 8-elem runs at K = k0+8h and k0+16+8h.
static __device__ __forceinline__ Frag load_bfrag(const unsigned short* colp, int k0, int hf) {
    Frag b;
    const unsigned short* p = colp + k0 + 8 * hf;
    b.u[0] = *(const uint4*)(p);
    b.u[1] = *(const uint4*)(p + 16);
    return b;
}

// ---------------------------------------------------------------------------
// f32 -> bf16 bulk convert (vectorized x4)
// ---------------------------------------------------------------------------
__global__ void cvt_f32_bf16(const float4* __restrict__ s, uint2* __restrict__ d, int n4) {
    int i = blockIdx.x * blockDim.x + threadIdx.x;
    if (i < n4) {
        float4 f = s[i];
        uint2 o;
        o.x = pack2(f.x, f.y);
        o.y = pack2(f.z, f.w);
        d[i] = o;
    }
}

// ---------------------------------------------------------------------------
// dic_z f32 [K=4096, E=2048]  ->  dicT bf16 [E=2048, K=4096]
// ---------------------------------------------------------------------------
__global__ void transpose_cvt(const float* __restrict__ dic, unsigned short* __restrict__ dicT) {
    __shared__ float t[32][33];
    int e0 = blockIdx.x * 32, k0 = blockIdx.y * 32;
    int tx = threadIdx.x, ty = threadIdx.y;   // (32, 8)
#pragma unroll
    for (int i = 0; i < 4; ++i)
        t[ty + 8 * i][tx] = dic[(size_t)(k0 + ty + 8 * i) * 2048 + e0 + tx];
    __syncthreads();
#pragma unroll
    for (int i = 0; i < 4; ++i)
        dicT[(size_t)(e0 + ty + 8 * i) * 4096 + k0 + tx] = f2bf(t[tx][ty + 8 * i]);
}

// ---------------------------------------------------------------------------
// Generic NT GEMM: C[M,N](bf16) = A[M,K](bf16) @ B[N,K](bf16)^T + bias(f32)
// 256 threads = 8 waves (wave32). Block tile 64(M) x 256(N); wave tile 32x64.
// ---------------------------------------------------------------------------
__global__ void __launch_bounds__(256)
gemm_nt_bf16(const unsigned short* __restrict__ A, const unsigned short* __restrict__ Bm,
             const float* __restrict__ bias, unsigned short* __restrict__ C,
             int Ndim, int Kdim) {
    const int wid = threadIdx.x >> 5, lane = threadIdx.x & 31;
    const int wm = wid >> 2, wn = wid & 3;
    const int m0 = blockIdx.y * 64 + wm * 32;
    const int n0 = blockIdx.x * 256 + wn * 64;
    const int r15 = lane & 15, hf = lane >> 4;

    v8f acc[2][4] = {};
    const unsigned short* Arow0 = A + (size_t)(m0 + r15) * Kdim;
    const unsigned short* Arow1 = Arow0 + (size_t)16 * Kdim;

    for (int kk = 0; kk < Kdim; kk += 32) {
        Frag a0 = load_afrag(Arow0, kk, hf);
        Frag a1 = load_afrag(Arow1, kk, hf);
#pragma unroll
        for (int j = 0; j < 4; ++j) {
            Frag b = load_bfrag(Bm + (size_t)(n0 + j * 16 + r15) * Kdim, kk, hf);
            acc[0][j] = wmma_bf16(a0, b, acc[0][j]);
            acc[1][j] = wmma_bf16(a1, b, acc[1][j]);
        }
    }
#pragma unroll
    for (int i = 0; i < 2; ++i)
#pragma unroll
        for (int j = 0; j < 4; ++j) {
            int n = n0 + j * 16 + r15;
            float bv = bias[n];
#pragma unroll
            for (int d = 0; d < 8; ++d) {
                int row = m0 + i * 16 + d + 8 * hf;
                C[(size_t)row * Ndim + n] = f2bf(acc[i][j][d] + bv);
            }
        }
}

// ---------------------------------------------------------------------------
// Fused attention: per block of 32 rows of h:
//   stage 0: async-copy prior (16KB) into LDS, overlapped with stage 1
//   stage 1: logits bf16 [32, 4096] -> padded LDS via WMMA (h @ dz^T / 32)
//   stage 2: in-place softmax * prior; wave-private rows, uint4 LDS sweeps,
//            __shfl_xor cross-lane reductions (no barriers between passes)
//   stage 3: z[32,2048] = W @ dicT^T via WMMA, A-fragments from LDS
// 8 waves: stage1 splits K (512 cols/wave), stage3 splits E (256 cols/wave).
// ---------------------------------------------------------------------------
#define ATT_K  4096
#define ATT_LD 4104   // padded row stride (bf16 elems): 8208B, spreads banks
__global__ void __launch_bounds__(256)
fused_attn(const unsigned short* __restrict__ hbf, const unsigned short* __restrict__ dzb,
           const unsigned short* __restrict__ dicT, const float* __restrict__ prior,
           float* __restrict__ out) {
    extern __shared__ unsigned short Wb[];   // [32][ATT_LD] bf16 logits -> weights
    __shared__ float priorS[ATT_K];          // 16KB staged copy of prior

    const int tid = threadIdx.x, wid = tid >> 5, lane = tid & 31;
    const int r15 = lane & 15, hf = lane >> 4;
    const int row0 = blockIdx.x * 32;

    // ---- Stage 0: prior -> LDS (async DMA if available; overlaps stage 1) ----
    {
        const float* gp = prior + tid * 16;
        float* lp = priorS + tid * 16;
#if HAVE_ASYNC_LDS
        // builtin expects (int4 addrspace(1)*, int4 addrspace(3)*, imm, imm)
        typedef __attribute__((address_space(1))) vi4* gvp_t;
        typedef __attribute__((address_space(3))) vi4* lvp_t;
#pragma unroll
        for (int j = 0; j < 4; ++j)
            __builtin_amdgcn_global_load_async_to_lds_b128(
                (gvp_t)(gp + j * 4), (lvp_t)(lp + j * 4), 0, 0);
#else
#pragma unroll
        for (int j = 0; j < 4; ++j)
            ((float4*)lp)[j] = ((const float4*)gp)[j];
#endif
    }

    // ---- Stage 1: logits = h_block @ dz^T, scaled by 1/sqrt(EMB)=1/32 ----
    {
        const unsigned short* A0 = hbf + (size_t)(row0 + r15) * 1024;
        const unsigned short* A1 = A0 + (size_t)16 * 1024;
        const int kw0 = wid * 512;
        for (int kt = 0; kt < 32; ++kt) {
            const int ktile = kw0 + kt * 16;
            const unsigned short* Brow = dzb + (size_t)(ktile + r15) * 1024;
            v8f acc0 = {}, acc1 = {};
            for (int dd = 0; dd < 1024; dd += 32) {
                Frag a0 = load_afrag(A0, dd, hf);
                Frag a1 = load_afrag(A1, dd, hf);
                Frag b  = load_bfrag(Brow, dd, hf);
                acc0 = wmma_bf16(a0, b, acc0);
                acc1 = wmma_bf16(a1, b, acc1);
            }
            const float sc = 0.03125f;   // 1/sqrt(1024)
            const int kk = ktile + r15;
#pragma unroll
            for (int d = 0; d < 8; ++d) {
                int rr = d + 8 * hf;
                Wb[rr * ATT_LD + kk]        = f2bf(acc0[d] * sc);
                Wb[(rr + 16) * ATT_LD + kk] = f2bf(acc1[d] * sc);
            }
        }
    }
#if HAVE_ASYNC_LDS
#if __has_builtin(__builtin_amdgcn_s_wait_asynccnt)
    __builtin_amdgcn_s_wait_asynccnt(0);
#else
    asm volatile("s_wait_asynccnt 0x0" ::: "memory");
#endif
#endif
    __syncthreads();

    // ---- Stage 2: in-place softmax * prior. Wave w owns rows 4w..4w+3. ----
    // Lane sweeps chunks (lane + 32*i): consecutive lanes -> consecutive 16B,
    // conflict-free b128 LDS traffic; reductions via __shfl_xor (wave32).
    {
        const int wrow0 = wid * 4;
        for (int rj = 0; rj < 4; ++rj) {
            uint4* Lv = (uint4*)(Wb + (wrow0 + rj) * ATT_LD) + lane;   // stride 32 chunks
            float mx = -3.0e38f;
#pragma unroll 4
            for (int i = 0; i < 16; ++i) {
                uint4 q = Lv[i * 32];
                mx = fmaxf(mx, fmaxf(fmaxf(fmaxf(lo_bf(q.x), hi_bf(q.x)),
                                           fmaxf(lo_bf(q.y), hi_bf(q.y))),
                                     fmaxf(fmaxf(lo_bf(q.z), hi_bf(q.z)),
                                           fmaxf(lo_bf(q.w), hi_bf(q.w)))));
            }
#pragma unroll
            for (int o = 16; o >= 1; o >>= 1) mx = fmaxf(mx, __shfl_xor(mx, o, 32));

            float s = 0.f;
#pragma unroll 4
            for (int i = 0; i < 16; ++i) {
                uint4 q = Lv[i * 32];
                float e0 = __expf(lo_bf(q.x) - mx), e1 = __expf(hi_bf(q.x) - mx);
                float e2 = __expf(lo_bf(q.y) - mx), e3 = __expf(hi_bf(q.y) - mx);
                float e4 = __expf(lo_bf(q.z) - mx), e5 = __expf(hi_bf(q.z) - mx);
                float e6 = __expf(lo_bf(q.w) - mx), e7 = __expf(hi_bf(q.w) - mx);
                s += ((e0 + e1) + (e2 + e3)) + ((e4 + e5) + (e6 + e7));
                q.x = pack2(e0, e1); q.y = pack2(e2, e3);
                q.z = pack2(e4, e5); q.w = pack2(e6, e7);
                Lv[i * 32] = q;
            }
#pragma unroll
            for (int o = 16; o >= 1; o >>= 1) s += __shfl_xor(s, o, 32);
            const float inv = 1.0f / s;

            const float4* pv = (const float4*)priorS + 2 * lane;   // k = chunk*8
#pragma unroll 4
            for (int i = 0; i < 16; ++i) {
                uint4 q = Lv[i * 32];
                float4 p0 = pv[i * 64], p1 = pv[i * 64 + 1];
                q.x = pack2(lo_bf(q.x) * p0.x * inv, hi_bf(q.x) * p0.y * inv);
                q.y = pack2(lo_bf(q.y) * p0.z * inv, hi_bf(q.y) * p0.w * inv);
                q.z = pack2(lo_bf(q.z) * p1.x * inv, hi_bf(q.z) * p1.y * inv);
                q.w = pack2(lo_bf(q.w) * p1.z * inv, hi_bf(q.w) * p1.w * inv);
                Lv[i * 32] = q;
            }
        }
    }
    __syncthreads();

    // ---- Stage 3: z_block = W(LDS) @ dicT^T ----
    {
        const int ew0 = wid * 256;
        const unsigned short* W0 = Wb + r15 * ATT_LD;        // padded: conflict-free
        const unsigned short* W1 = W0 + 16 * ATT_LD;
        for (int et = 0; et < 16; ++et) {
            const int etile = ew0 + et * 16;
            const unsigned short* Brow = dicT + (size_t)(etile + r15) * 4096;
            // hint next tile's stream into cache (global_prefetch_b8)
            __builtin_prefetch(dicT + (size_t)(etile + 16 + r15) * 4096, 0, 0);
            v8f acc0 = {}, acc1 = {};
            for (int kk = 0; kk < 4096; kk += 32) {
                Frag a0 = load_afrag(W0, kk, hf);            // ds_read_b128 x2
                Frag a1 = load_afrag(W1, kk, hf);
                Frag b  = load_bfrag(Brow, kk, hf);
                acc0 = wmma_bf16(a0, b, acc0);
                acc1 = wmma_bf16(a1, b, acc1);
            }
            const int e = etile + r15;
#pragma unroll
            for (int d = 0; d < 8; ++d) {
                out[(size_t)(row0 + d + 8 * hf) * 2048 + e]      = acc0[d];
                out[(size_t)(row0 + 16 + d + 8 * hf) * 2048 + e] = acc1[d];
            }
        }
    }
}

// ---------------------------------------------------------------------------
// Host side
// ---------------------------------------------------------------------------
extern "C" void kernel_launch(void* const* d_in, const int* in_sizes, int n_in,
                              void* d_out, int out_size, void* d_ws, size_t ws_size,
                              hipStream_t stream) {
    const float* y     = (const float*)d_in[0];   // [64,256,1024]
    const float* Wy_w  = (const float*)d_in[1];   // [1024,1024]
    const float* Wy_b  = (const float*)d_in[2];   // [1024]
    const float* Wz_w  = (const float*)d_in[3];   // [1024,2048]
    const float* Wz_b  = (const float*)d_in[4];   // [1024]
    const float* dic   = (const float*)d_in[5];   // [4096,2048]
    const float* prior = (const float*)d_in[6];   // [4096]
    float* out = (float*)d_out;                   // [64,256,2048] f32

    char* ws = (char*)d_ws;                       // needs ~110 MB
    const size_t MB = (size_t)1 << 20;
    unsigned short* yb   = (unsigned short*)(ws);             // 32 MB  [16384,1024]
    unsigned short* hbf  = (unsigned short*)(ws + 32  * MB);  // 32 MB  [16384,1024]
    unsigned short* dicb = (unsigned short*)(ws + 64  * MB);  // 16 MB  [4096,2048]
    unsigned short* dicT = (unsigned short*)(ws + 80  * MB);  // 16 MB  [2048,4096]
    unsigned short* dzb  = (unsigned short*)(ws + 96  * MB);  //  8 MB  [4096,1024]
    unsigned short* Wyb  = (unsigned short*)(ws + 104 * MB);  //  2 MB  [1024,1024]
    unsigned short* Wzb  = (unsigned short*)(ws + 106 * MB);  //  4 MB  [1024,2048]

    auto cvt = [&](const float* s, unsigned short* d, int n) {
        int n4 = n / 4;
        cvt_f32_bf16<<<(n4 + 255) / 256, 256, 0, stream>>>((const float4*)s, (uint2*)d, n4);
    };
    cvt(y,    yb,   16384 * 1024);
    cvt(Wy_w, Wyb,  1024 * 1024);
    cvt(Wz_w, Wzb,  1024 * 2048);
    cvt(dic,  dicb, 4096 * 2048);
    transpose_cvt<<<dim3(64, 128), dim3(32, 8), 0, stream>>>(dic, dicT);

    // h = y @ Wy^T + by : [16384,1024], K=1024
    gemm_nt_bf16<<<dim3(4, 256), 256, 0, stream>>>(yb, Wyb, Wy_b, hbf, 1024, 1024);
    // dz = dic @ Wz^T + bz : [4096,1024], K=2048
    gemm_nt_bf16<<<dim3(4, 64), 256, 0, stream>>>(dicb, Wzb, Wz_b, dzb, 1024, 2048);

    // fused attention: 512 blocks x 32 rows, padded 256.5KB dynamic LDS
    const int smem = 32 * ATT_LD * (int)sizeof(unsigned short);   // 262656
    (void)hipFuncSetAttribute((const void*)fused_attn,
                              hipFuncAttributeMaxDynamicSharedMemorySize, smem);
    fused_attn<<<512, 256, smem, stream>>>(hbf, dzb, dicT, prior, out);
}
